// MCC_module_15865609191814
// MI455X (gfx1250) — compile-verified, matrix operand order checked by reading, not run on hardware
//
#include <hip/hip_runtime.h>
#include <hip/hip_bf16.h>
#include <math.h>

typedef unsigned short u16;
typedef __attribute__((ext_vector_type(8)))  u16    u16x8;
typedef __attribute__((ext_vector_type(8)))  __bf16 bf16x8;
typedef __attribute__((ext_vector_type(16))) __bf16 bf16x16;
typedef __attribute__((ext_vector_type(8)))  float  v8f;

#define BDIM 32
#define CDIM 512
#define MDIM 3136   // 56*56
#define RDIM 64     // C / r

// ---------- helpers ----------
__device__ __forceinline__ u16 f2bf(float f) {
    unsigned x = __float_as_uint(f);
    unsigned r = x + 0x7FFFu + ((x >> 16) & 1u);   // round-to-nearest-even
    return (u16)(r >> 16);
}

// Load a 16x32 bf16 fragment in the CDNA5 "A" layout from a row-major matrix.
// lane l<16 : row = base_row + (l&15), K = k0+{0..7, 16..23}
// lane l>=16: row = base_row + (l&15), K = k0+{8..15, 24..31}
// Used for both A (rows of A) and B (rows of B; all our B operands are symmetric).
__device__ __forceinline__ bf16x16 load_frag(const u16* __restrict__ base,
                                             int row, int ld, int k0) {
    const int lane = threadIdx.x & 31;
    const int h = lane >> 4;
    const u16* p = base + (size_t)row * ld + k0 + h * 8;
    union { u16x8 u; bf16x8 b; } lo, hi;
    lo.u = *(const u16x8*)(p);
    hi.u = *(const u16x8*)(p + 16);
    bf16x16 out;
#pragma unroll
    for (int i = 0; i < 8; ++i) { out[i] = lo.b[i]; out[8 + i] = hi.b[i]; }
    return out;
}

// ---------- batched WMMA GEMM: D = post(A @ B), A,B bf16 row-major, ld = K ----------
// Wave tile 64x64 (16 WMMA accum tiles); workgroup = 8 waves (2x4) -> 128x256 tile.
#define GEMM_GRAM  0   // Df32 = (A@B) / K
#define GEMM_COPY  1   // Dbf  = A@B
#define GEMM_NS    2   // Dbf  = 0.5*(3I - A@B)
#define GEMM_FINAL 3   // Df32 = (A@B) * sqrt(normA[b])

__global__ __launch_bounds__(256)
void k_gemm(const u16* __restrict__ A, const u16* __restrict__ B,
            u16* __restrict__ Dbf, float* __restrict__ Df32,
            int N, int K, long long strideA, long long strideB,
            int mode, const float* __restrict__ normA)
{
    const int b = blockIdx.z;
    const u16* Ab = A + (long long)b * strideA;
    const u16* Bb = B + (long long)b * strideB;
    const int wave = threadIdx.x >> 5;
    const int lane = threadIdx.x & 31;
    const int wr = wave >> 2;                 // 0..1 : 64-row slab
    const int wc = wave & 3;                  // 0..3 : 64-col slab
    const int i0 = blockIdx.y * 128 + wr * 64;
    const int j0 = blockIdx.x * 256 + wc * 64;
    const int m15 = lane & 15;

    v8f acc[4][4];
#pragma unroll
    for (int r = 0; r < 4; ++r)
#pragma unroll
        for (int c = 0; c < 4; ++c)
            acc[r][c] = (v8f){0.f, 0.f, 0.f, 0.f, 0.f, 0.f, 0.f, 0.f};

    for (int k = 0; k < K; k += 32) {
        if (k + 32 < K) {  // gfx1250 global_prefetch path
            __builtin_prefetch(Ab + (size_t)(i0 + m15) * K + k + 32, 0, 1);
            __builtin_prefetch(Bb + (size_t)(j0 + m15) * K + k + 32, 0, 1);
        }
        bf16x16 afrag[4], bfrag[4];
#pragma unroll
        for (int r = 0; r < 4; ++r)
            afrag[r] = load_frag(Ab, i0 + 16 * r + m15, K, k);
#pragma unroll
        for (int c = 0; c < 4; ++c)
            bfrag[c] = load_frag(Bb, j0 + 16 * c + m15, K, k);
#pragma unroll
        for (int r = 0; r < 4; ++r)
#pragma unroll
            for (int c = 0; c < 4; ++c)
                acc[r][c] = __builtin_amdgcn_wmma_f32_16x16x32_bf16(
                    false, afrag[r], false, bfrag[c],
                    (short)0, acc[r][c], false, false);
    }

    const float kinv = 1.0f / (float)K;
    const float fsc  = (mode == GEMM_FINAL) ? sqrtf(normA[b]) : 0.f;
    const long long dbase = (long long)b * (long long)N * (long long)N;
    const int hh = (lane >> 4) * 8;
#pragma unroll
    for (int r = 0; r < 4; ++r) {
#pragma unroll
        for (int c = 0; c < 4; ++c) {
            const int col = j0 + 16 * c + m15;
#pragma unroll
            for (int rr = 0; rr < 8; ++rr) {
                const int row = i0 + 16 * r + rr + hh;   // C/D layout: VGPR rr, half hh
                const float a = acc[r][c][rr];
                const long long idx = dbase + (long long)row * N + col;
                if (mode == GEMM_GRAM)      Df32[idx] = a * kinv;
                else if (mode == GEMM_COPY) Dbf[idx]  = f2bf(a);
                else if (mode == GEMM_NS)   Dbf[idx]  = f2bf(0.5f * (((row == col) ? 3.0f : 0.0f) - a));
                else                        Df32[idx] = a * fsc;
            }
        }
    }
}

// ---------- per-(b,c) spatial mean + centered bf16 copy ----------
__global__ __launch_bounds__(256)
void k_mean_center(const float* __restrict__ x, float* __restrict__ pooled,
                   u16* __restrict__ xbar)
{
    const int bc = blockIdx.x;
    const float* row = x + (size_t)bc * MDIM;
    __shared__ float sdata[256];
    float s = 0.f;
    for (int i = threadIdx.x; i < MDIM / 4; i += 256) {
        float4 v = ((const float4*)row)[i];
        s += v.x + v.y + v.z + v.w;
    }
    sdata[threadIdx.x] = s;
    __syncthreads();
    for (int off = 128; off > 0; off >>= 1) {
        if (threadIdx.x < off) sdata[threadIdx.x] += sdata[threadIdx.x + off];
        __syncthreads();
    }
    const float mean = sdata[0] * (1.0f / (float)MDIM);
    if (threadIdx.x == 0) pooled[bc] = mean;
    u16* orow = xbar + (size_t)bc * MDIM;
    for (int i = threadIdx.x; i < MDIM; i += 256)
        orow[i] = f2bf(row[i] - mean);
}

// ---------- trace norm + An(bf16) + ZY0(bf16) ----------
__global__ __launch_bounds__(256)
void k_trace_scale(const float* __restrict__ cov, u16* __restrict__ An,
                   u16* __restrict__ Z, float* __restrict__ normA)
{
    const int b = blockIdx.x;
    const float* Cb = cov + (size_t)b * CDIM * CDIM;
    __shared__ float sdata[256];
    float s = 0.f;
    for (int i = threadIdx.x; i < CDIM; i += 256) s += Cb[(size_t)i * CDIM + i];
    sdata[threadIdx.x] = s;
    __syncthreads();
    for (int off = 128; off > 0; off >>= 1) {
        if (threadIdx.x < off) sdata[threadIdx.x] += sdata[threadIdx.x + off];
        __syncthreads();
    }
    const float norm = sdata[0];
    const float inv = 1.0f / norm;
    if (threadIdx.x == 0) normA[b] = norm;
    u16* Ab = An + (size_t)b * CDIM * CDIM;
    u16* Zb = Z  + (size_t)b * CDIM * CDIM;
    for (int i = threadIdx.x; i < CDIM * CDIM; i += 256) {
        const int rrow = i >> 9, ccol = i & 511;
        const float an = Cb[i] * inv;
        Ab[i] = f2bf(an);
        Zb[i] = f2bf(0.5f * (((rrow == ccol) ? 3.0f : 0.0f) - an));
    }
}

// ---------- cov_sum[b][n] = mean_m cov_sqrt[b][m][n] ----------
__global__ void k_rowmean(const float* __restrict__ covs, float* __restrict__ out)
{
    const int t = blockIdx.x * blockDim.x + threadIdx.x;   // B*C threads
    const int b = t >> 9, n = t & 511;
    const float* Cb = covs + (size_t)b * CDIM * CDIM;
    float s = 0.f;
    for (int m = 0; m < CDIM; ++m) s += Cb[(size_t)m * CDIM + n];
    out[t] = s * (1.0f / (float)CDIM);
}

// ---------- SE branch: sigmoid(W2 relu(W1 v + b1) + b2) ----------
__global__ __launch_bounds__(256)
void k_se(const float* __restrict__ vin, const float* __restrict__ w1,
          const float* __restrict__ b1, const float* __restrict__ w2,
          const float* __restrict__ b2, float* __restrict__ scale)
{
    const int b = blockIdx.x;
    __shared__ float v[CDIM];
    __shared__ float h[RDIM];
    for (int i = threadIdx.x; i < CDIM; i += 256) v[i] = vin[b * CDIM + i];
    __syncthreads();
    if (threadIdx.x < RDIM) {
        const int r = threadIdx.x;
        float s = b1[r];
        for (int c = 0; c < CDIM; ++c) s += w1[r * CDIM + c] * v[c];
        h[r] = fmaxf(s, 0.f);
    }
    __syncthreads();
    for (int c = threadIdx.x; c < CDIM; c += 256) {
        float s = b2[c];
        for (int r = 0; r < RDIM; ++r) s += w2[c * RDIM + r] * h[r];
        scale[b * CDIM + c] = 1.0f / (1.0f + __expf(-s));
    }
}

// ---------- fused scale + BN(eval) + ReLU + residual ----------
__global__ void k_final(const float* __restrict__ x,
                        const float* __restrict__ fccs, const float* __restrict__ sccs,
                        const float* __restrict__ gamma, const float* __restrict__ beta,
                        const float* __restrict__ bmean, const float* __restrict__ bvar,
                        float* __restrict__ out)
{
    const long long q = (long long)blockIdx.x * blockDim.x + threadIdx.x; // float4 index
    const long long e0 = q * 4;
    const int bc = (int)(e0 / MDIM);       // MDIM divisible by 4 -> no straddle
    const int c = bc & 511;
    const float s   = fccs[bc] + sccs[bc];
    const float inv = gamma[c] * rsqrtf(bvar[c] + 1e-5f);
    const float mu = bmean[c], bt = beta[c];
    float4 xv = ((const float4*)x)[q];
    float4 o;
    o.x = fmaxf((s * xv.x - mu) * inv + bt, 0.f) + xv.x;
    o.y = fmaxf((s * xv.y - mu) * inv + bt, 0.f) + xv.y;
    o.z = fmaxf((s * xv.z - mu) * inv + bt, 0.f) + xv.z;
    o.w = fmaxf((s * xv.w - mu) * inv + bt, 0.f) + xv.w;
    ((float4*)out)[q] = o;
}

// ---------- host ----------
extern "C" void kernel_launch(void* const* d_in, const int* in_sizes, int n_in,
                              void* d_out, int out_size, void* d_ws, size_t ws_size,
                              hipStream_t stream)
{
    const float* x       = (const float*)d_in[0];
    const float* fcc_w1  = (const float*)d_in[1];
    const float* fcc_b1  = (const float*)d_in[2];
    const float* fcc_w2  = (const float*)d_in[3];
    const float* fcc_b2  = (const float*)d_in[4];
    const float* du_w1   = (const float*)d_in[5];
    const float* du_b1   = (const float*)d_in[6];
    const float* du_w2   = (const float*)d_in[7];
    const float* du_b2   = (const float*)d_in[8];
    const float* bn_g    = (const float*)d_in[9];
    const float* bn_b    = (const float*)d_in[10];
    const float* bn_m    = (const float*)d_in[11];
    const float* bn_v    = (const float*)d_in[12];
    (void)in_sizes; (void)n_in; (void)out_size; (void)ws_size;

    char* ws = (char*)d_ws;
    size_t off = 0;
    auto alloc = [&](size_t bytes) {
        void* p = ws + off;
        off += (bytes + 255) & ~(size_t)255;
        return p;
    };

    const size_t BC  = (size_t)BDIM * CDIM;
    const size_t BCC = (size_t)BDIM * CDIM * CDIM;
    const size_t BCM = (size_t)BDIM * CDIM * MDIM;

    float* pooled  = (float*)alloc(BC * 4);
    float* covsum  = (float*)alloc(BC * 4);
    float* fccs    = (float*)alloc(BC * 4);
    float* sccs    = (float*)alloc(BC * 4);
    float* normA   = (float*)alloc(BDIM * 4);
    u16*   xbar    = (u16*)  alloc(BCM * 2);
    float* covf32  = (float*)alloc(BCC * 4);
    u16*   buf[4];
    for (int i = 0; i < 4; ++i) buf[i] = (u16*)alloc(BCC * 2);

    const long long sCC = (long long)CDIM * CDIM;
    const long long sCM = (long long)CDIM * MDIM;
    const dim3 gemmGrid(2, 4, BDIM);   // 512/256 x 512/128 x batches

    // 1) spatial means + centered bf16 copy of x
    k_mean_center<<<BDIM * CDIM, 256, 0, stream>>>(x, pooled, xbar);

    // 2) cov = (1/M) xbar xbar^T  (batched WMMA Gram)
    k_gemm<<<gemmGrid, 256, 0, stream>>>(xbar, xbar, nullptr, covf32,
                                         CDIM, MDIM, sCM, sCM, GEMM_GRAM, nullptr);

    // 3) normA = trace; An = cov/normA (buf0); Z = 0.5*(3I - An) (buf1)
    k_trace_scale<<<BDIM, 256, 0, stream>>>(covf32, buf[0], buf[1], normA);

    // 4) Y = An @ Z  -> buf2
    k_gemm<<<gemmGrid, 256, 0, stream>>>(buf[0], buf[1], buf[2], nullptr,
                                         CDIM, CDIM, sCC, sCC, GEMM_COPY, nullptr);

    // 5) three Newton-Schulz refinement steps (all operands symmetric)
    int Yb = 2, Zb = 1, Fb = 0;        // buf3 is always T
    for (int it = 0; it < 3; ++it) {
        k_gemm<<<gemmGrid, 256, 0, stream>>>(buf[Zb], buf[Yb], buf[3], nullptr,
                                             CDIM, CDIM, sCC, sCC, GEMM_NS, nullptr);
        k_gemm<<<gemmGrid, 256, 0, stream>>>(buf[Yb], buf[3], buf[Fb], nullptr,
                                             CDIM, CDIM, sCC, sCC, GEMM_COPY, nullptr);
        k_gemm<<<gemmGrid, 256, 0, stream>>>(buf[3], buf[Zb], buf[Yb], nullptr,
                                             CDIM, CDIM, sCC, sCC, GEMM_COPY, nullptr);
        const int nY = Fb, nZ = Yb, nF = Zb;
        Yb = nY; Zb = nZ; Fb = nF;
    }

    // 6) final: T = 0.5*(3I - Z@Y); cov_sqrt = (Y@T) * sqrt(normA)
    k_gemm<<<gemmGrid, 256, 0, stream>>>(buf[Zb], buf[Yb], buf[3], nullptr,
                                         CDIM, CDIM, sCC, sCC, GEMM_NS, nullptr);
    k_gemm<<<gemmGrid, 256, 0, stream>>>(buf[Yb], buf[3], nullptr, covf32,
                                         CDIM, CDIM, sCC, sCC, GEMM_FINAL, normA);

    // 7) cov_sum = mean over rows
    k_rowmean<<<(BDIM * CDIM) / 256, 256, 0, stream>>>(covf32, covsum);

    // 8) SE branches
    k_se<<<BDIM, 256, 0, stream>>>(pooled, fcc_w1, fcc_b1, fcc_w2, fcc_b2, fccs);
    k_se<<<BDIM, 256, 0, stream>>>(covsum, du_w1, du_b1, du_w2, du_b2, sccs);

    // 9) fused (fcc+scc)*x -> BN -> ReLU -> +x
    const long long q4 = (long long)BDIM * CDIM * MDIM / 4;
    k_final<<<(unsigned)(q4 / 256), 256, 0, stream>>>(x, fccs, sccs, bn_g, bn_b,
                                                      bn_m, bn_v, (float*)d_out);
}
